// TRMPolicy_10685878632865
// MI455X (gfx1250) — matrix operand order repaired and checked by reading, not compiled, not run on hardware
//
#include <hip/hip_runtime.h>
#include <hip/hip_bf16.h>
#include <math.h>

typedef __attribute__((ext_vector_type(16))) _Float16 v16h;
typedef __attribute__((ext_vector_type(8)))  _Float16 v8h;
typedef __attribute__((ext_vector_type(4)))  _Float16 v4h;
typedef __attribute__((ext_vector_type(8)))  float    v8f;

// Wave-local LDS RAW fence using CDNA5 split dependency counters.
static __device__ __forceinline__ void lds_fence() {
#if __has_builtin(__builtin_amdgcn_s_wait_dscnt)
  __builtin_amdgcn_s_wait_dscnt(0);
  asm volatile("" ::: "memory");
#else
  asm volatile("s_wait_dscnt 0" ::: "memory");
#endif
}

// Convert an (Nreal x Kreal) row-major fp32 weight (used as inp @ W^T) into
// f16 B-matrix WMMA fragments in LDS, zero-padded to KT*32 x NT*16.
// Fragment (kt,nt) occupies 512 halfs; lane l holds 16 halfs at l*16:
//   n = nt*16 + (l & 15),  k = kt*32 + (l>>4)*8 + e + (e>=8 ? 8 : 0)     (e = 0..15)
static __device__ void build_w(const float* __restrict__ W, _Float16* dst,
                               int Kreal, int Nreal, int KT, int NT) {
  const int total = KT * NT * 512;
  for (int i = threadIdx.x; i < total; i += blockDim.x) {
    int frag = i >> 9;
    int r    = i & 511;
    int lane = r >> 4;
    int e    = r & 15;
    int kt = frag / NT, nt = frag - kt * NT;
    int n = nt * 16 + (lane & 15);
    int k = kt * 32 + ((lane >> 4) << 3) + e + ((e >> 3) << 3);
    float v = (k < Kreal && n < Nreal) ? W[n * Kreal + k] : 0.0f;
    dst[i] = (_Float16)v;
  }
}

// 32-row GEMM (two 16-row M blocks sharing each B fragment):
//   O(32 x NT*16) = relu?(A(32 rows, gathered k-tiles) * Wf + bias)
// koff[kt] gives the starting column (in halfs) of k-tile kt inside A,
// letting callers gather non-contiguous input slices (e.g. [z | y-pad]).
// Accumulators start at zero (first WMMA can use inline C=0); bias is
// added at writeback, fused ahead of the relu max.
template<int KT, int NT, bool RELU>
static __device__ __forceinline__ void mm_tile32(
    const _Float16* __restrict__ A, int lda, const int (&koff)[KT],
    const _Float16* __restrict__ Wf,
    const float* __restrict__ bias,
    _Float16* __restrict__ O, int ldo,
    int lane) {
  const int c0 = lane & 15;   // A row-in-block / C-D column
  const int hh = lane >> 4;
  v8f acc[2][NT] = {};
#pragma unroll
  for (int kt = 0; kt < KT; ++kt) {
    union { v16h v; v8h p[2]; } a0, a1;
    const _Float16* ap0 = A + c0 * lda + koff[kt] + hh * 8;
    const _Float16* ap1 = ap0 + 16 * lda;
    a0.p[0] = *(const v8h*)(ap0);
    a0.p[1] = *(const v8h*)(ap0 + 16);
    a1.p[0] = *(const v8h*)(ap1);
    a1.p[1] = *(const v8h*)(ap1 + 16);
#pragma unroll
    for (int nt = 0; nt < NT; ++nt) {
      v16h b = *(const v16h*)(Wf + (kt * NT + nt) * 512 + lane * 16);
      acc[0][nt] = __builtin_amdgcn_wmma_f32_16x16x32_f16(
          false, a0.v, false, b, (short)0, acc[0][nt], false, false);
      acc[1][nt] = __builtin_amdgcn_wmma_f32_16x16x32_f16(
          false, a1.v, false, b, (short)0, acc[1][nt], false, false);
    }
  }
  // D layout: lane holds column c0, rows m = rb*16 + r + 8*hh.
#pragma unroll
  for (int nt = 0; nt < NT; ++nt) {
    float b = bias[nt * 16 + c0];
#pragma unroll
    for (int rb = 0; rb < 2; ++rb) {
#pragma unroll
      for (int r = 0; r < 8; ++r) {
        float v = acc[rb][nt][r] + b;
        if (RELU) v = fmaxf(v, 0.0f);
        int mrow = rb * 16 + r + 8 * hh;
        O[mrow * ldo + nt * 16 + c0] = (_Float16)v;
      }
    }
  }
}

__global__ __launch_bounds__(64)
void trm_policy_kernel(const float* __restrict__ state,
                       const int* __restrict__ prev_action,
                       const float* __restrict__ z0,
                       const float* __restrict__ Wt1, const float* __restrict__ bt1,
                       const float* __restrict__ Wt2, const float* __restrict__ bt2,
                       const float* __restrict__ Wh1, const float* __restrict__ bh1,
                       const float* __restrict__ Wh2, const float* __restrict__ bh2,
                       const float* __restrict__ emb,
                       float* __restrict__ out, int batch) {
  // Weights pre-swizzled into WMMA B-fragment order (f16).
  __shared__ alignas(32) _Float16 wt1[16 * 512];  // (112->128) x 64
  __shared__ alignas(32) _Float16 wt2[4 * 512];   // 64 x 32
  __shared__ alignas(32) _Float16 wh1[4 * 512];   // (48->64) x 32
  __shared__ alignas(32) _Float16 wh2[512];       // 32 x (4->16)
  // Per-wave activation tiles (32 rows per wave).
  // act = [ z(32) | x(64) | y(16) | 0(16) ]: serves tiny input (k-tiles
  // 0,32,64,96) AND head input (k-tiles 0,96 = [z | y,0]).
  __shared__ alignas(32) _Float16 act[2][32 * 128];
  __shared__ alignas(32) _Float16 hbuf[2][32 * 64];  // hidden scratch
  __shared__ alignas(16) float    lbuf[2][32 * 4];   // logits
  __shared__ float bt1s[64], bt2s[32], bh1s[32], bh2s[16], embs[64];
  __shared__ alignas(8) _Float16 z0h[32];

  build_w(Wt1, wt1, 112, 64, 4, 4);
  build_w(Wt2, wt2, 64, 32, 2, 2);
  build_w(Wh1, wh1, 48, 32, 2, 2);
  build_w(Wh2, wh2, 32, 4, 1, 1);
  {
    int t = threadIdx.x;
    if (t < 64) { bt1s[t] = bt1[t]; embs[t] = emb[t]; }
    if (t < 32) { bt2s[t] = bt2[t]; bh1s[t] = bh1[t]; z0h[t] = (_Float16)z0[t]; }
    if (t < 16) { bh2s[t] = (t < 4) ? bh2[t] : 0.0f; }
  }
  __syncthreads();

  const int lane = threadIdx.x & 31;
  const int w    = threadIdx.x >> 5;
  const int m    = lane & 15;
  const int h    = lane >> 4;
  _Float16* actw  = act[w];
  _Float16* hbufw = hbuf[w];
  float*    lbufw = lbuf[w];

  constexpr int K_TINY1[4] = {0, 32, 64, 96};  // z | x | y,pad
  constexpr int K_TINY2[2] = {0, 32};          // hbuf h(64)
  constexpr int K_HEAD1[2] = {0, 96};          // z | y,pad (from act!)

  const int ntiles = batch >> 5;   // 32 rows per tile

  for (int tile = blockIdx.x * 2 + w; tile < ntiles; tile += gridDim.x * 2) {
    const int R = tile << 5;

    // ---- stage inputs: lane owns row `lane` of the 32-row tile ----
    {
      const float* xr = state + (size_t)(R + lane) * 64;
      _Float16* arow = actw + lane * 128;
#pragma unroll
      for (int i = 0; i < 64; i += 4) {
        float4 v = *(const float4*)(xr + i);
        v4h hv = { (_Float16)v.x, (_Float16)v.y, (_Float16)v.z, (_Float16)v.w };
        *(v4h*)(arow + 32 + i) = hv;
      }
      const int nt2 = tile + gridDim.x * 2;
      if (nt2 < ntiles)
        __builtin_prefetch(state + ((size_t)nt2 * 32 + lane) * 64, 0, 1);
      // z0 -> act[:, 0:32)
#pragma unroll
      for (int i = 0; i < 32; i += 4)
        *(v4h*)(arow + i) = *(const v4h*)(z0h + i);
      // y = emb[prev_action] -> act[:, 96:112)
      int a = prev_action[R + lane];
#pragma unroll
      for (int i = 0; i < 16; ++i)
        arow[96 + i] = (_Float16)embs[a * 16 + i];
      // zero pad act[:, 112:128)
      v4h zz = { (_Float16)0.0f, (_Float16)0.0f, (_Float16)0.0f, (_Float16)0.0f };
#pragma unroll
      for (int i = 0; i < 16; i += 4) *(v4h*)(arow + 112 + i) = zz;
    }
    lds_fence();

#pragma unroll 1
    for (int sup = 0; sup < 2; ++sup) {
#pragma unroll 1
      for (int inner = 0; inner < 2; ++inner) {
        // tiny L1: (32x112) -> relu -> hbuf(32x64)
        mm_tile32<4, 4, true>(actw, 128, K_TINY1, wt1, bt1s, hbufw, 64, lane);
        lds_fence();
        // tiny L2: (32x64) -> z -> act[:, :32]
        mm_tile32<2, 2, false>(hbufw, 64, K_TINY2, wt2, bt2s, actw, 128, lane);
        lds_fence();
      }
      // head L1: (32x[z|y]) -> relu -> hbuf[:, :32]   (reads act directly)
      mm_tile32<2, 2, true>(actw, 128, K_HEAD1, wh1, bh1s, hbufw, 64, lane);
      lds_fence();
      // head L2: (32x32) x Wh2(32x4, padded to 16) -> logits (B frag shared)
      {
        union { v16h v; v8h p[2]; } a0, a1;
        const _Float16* ap0 = hbufw + m * 64 + h * 8;
        const _Float16* ap1 = ap0 + 16 * 64;
        a0.p[0] = *(const v8h*)(ap0);
        a0.p[1] = *(const v8h*)(ap0 + 16);
        a1.p[0] = *(const v8h*)(ap1);
        a1.p[1] = *(const v8h*)(ap1 + 16);
        v16h b = *(const v16h*)(wh2 + lane * 16);
        v8f c0 = {}, c1 = {};
        c0 = __builtin_amdgcn_wmma_f32_16x16x32_f16(false, a0.v, false, b,
                                                    (short)0, c0, false, false);
        c1 = __builtin_amdgcn_wmma_f32_16x16x32_f16(false, a1.v, false, b,
                                                    (short)0, c1, false, false);
        if (m < 4) {
          float bb = bh2s[m];
#pragma unroll
          for (int r = 0; r < 8; ++r) {
            lbufw[(r + 8 * h) * 4 + m]      = c0[r] + bb;
            lbufw[(16 + r + 8 * h) * 4 + m] = c1[r] + bb;
          }
        }
      }
      lds_fence();

      if (sup == 0) {
        // y = softmax(logits) @ emb ; lane owns row `lane` (fp32 math).
        float l0 = lbufw[lane * 4 + 0], l1 = lbufw[lane * 4 + 1];
        float l2 = lbufw[lane * 4 + 2], l3 = lbufw[lane * 4 + 3];
        float mx = fmaxf(fmaxf(l0, l1), fmaxf(l2, l3));
        float e0 = __expf(l0 - mx), e1 = __expf(l1 - mx);
        float e2 = __expf(l2 - mx), e3 = __expf(l3 - mx);
        float inv = 1.0f / (e0 + e1 + e2 + e3);
        e0 *= inv; e1 *= inv; e2 *= inv; e3 *= inv;
        _Float16* arow = actw + lane * 128;
#pragma unroll
        for (int j = 0; j < 16; ++j) {
          float yv = e0 * embs[j] + e1 * embs[16 + j] +
                     e2 * embs[32 + j] + e3 * embs[48 + j];
          arow[96 + j] = (_Float16)yv;
        }
        lds_fence();
      } else {
        float4 o = *(const float4*)(lbufw + lane * 4);
        *(float4*)(out + (size_t)(R + lane) * 4) = o;
      }
    }
  }
}

extern "C" void kernel_launch(void* const* d_in, const int* in_sizes, int n_in,
                              void* d_out, int out_size, void* d_ws, size_t ws_size,
                              hipStream_t stream) {
  const float* state = (const float*)d_in[0];
  const int*   pa    = (const int*)d_in[1];
  const float* z0    = (const float*)d_in[2];
  const float* Wt1   = (const float*)d_in[3];
  const float* bt1   = (const float*)d_in[4];
  const float* Wt2   = (const float*)d_in[5];
  const float* bt2   = (const float*)d_in[6];
  const float* Wh1   = (const float*)d_in[7];
  const float* bh1   = (const float*)d_in[8];
  const float* Wh2   = (const float*)d_in[9];
  const float* bh2   = (const float*)d_in[10];
  const float* emb   = (const float*)d_in[11];
  float* out = (float*)d_out;

  int batch  = in_sizes[0] / 64;       // X_DIM = 64
  int ntiles = batch / 32;             // 32 rows per wave-tile
  int blocks = (ntiles + 1) / 2;       // 2 waves per block
  if (blocks > 2048) blocks = 2048;
  if (blocks < 1) blocks = 1;

  trm_policy_kernel<<<blocks, 64, 0, stream>>>(state, pa, z0, Wt1, bt1, Wt2, bt2,
                                               Wh1, bh1, Wh2, bh2, emb, out, batch);
}
